// MultiBoxLoss_5325759447799
// MI455X (gfx1250) — compile-verified, matrix operand order checked by reading, not run on hardware
//
#include <hip/hip_runtime.h>

#define NCLS 81
#define B_ 64
#define P_ 25000
#define T_ 16
#define NTILES ((P_ + 15) / 16)   // 1563 16-row WMMA tiles per batch

typedef __attribute__((ext_vector_type(8)))  float          v8f;
typedef __attribute__((ext_vector_type(16))) __bf16         v16bf;
typedef __attribute__((ext_vector_type(16))) unsigned short v16us;
typedef __attribute__((ext_vector_type(4)))  unsigned int   u32x4;
typedef __attribute__((ext_vector_type(8)))  int            i32x8;
typedef __attribute__((ext_vector_type(4)))  int            i32x4;

__device__ __forceinline__ unsigned short f2bf_rne(float f) {
    unsigned u = __float_as_uint(f);
    unsigned r = u + 0x7FFFu + ((u >> 16) & 1u);
    return (unsigned short)(r >> 16);
}

__device__ __forceinline__ float sl1(float d) {
    float a = fabsf(d);
    return a < 1.0f ? 0.5f * d * d : a - 0.5f;
}

// ---------------- K0: zero the small accumulators ----------------
__global__ void k0_init(unsigned long long* bp, int* num_pos, float* negsum, float* scal) {
    int t = blockIdx.x * blockDim.x + threadIdx.x;
    if (t < B_ * T_) bp[t] = 0ull;
    if (t < B_) { num_pos[t] = 0; negsum[t] = 0.0f; }
    if (t < 2) scal[t] = 0.0f;
}

// ---------------- K1: IoU match (best truth per prior, best prior per truth) ----------------
__global__ void __launch_bounds__(256)
k1_match(const float* __restrict__ priors, const float* __restrict__ targets,
         float* __restrict__ bt_ov, int* __restrict__ bt_idx,
         unsigned long long* __restrict__ bp) {
    const int b = blockIdx.y;
    const int p = blockIdx.x * blockDim.x + threadIdx.x;
    __shared__ float st[T_ * 5];
    __shared__ unsigned long long sbp[T_];
    if (threadIdx.x < T_ * 5) st[threadIdx.x] = targets[b * T_ * 5 + threadIdx.x];
    if (threadIdx.x < T_)     sbp[threadIdx.x] = 0ull;
    __syncthreads();
    if (p < P_) {
        float cx = priors[p*4+0], cy = priors[p*4+1], w = priors[p*4+2], h = priors[p*4+3];
        float bx0 = cx - 0.5f*w, by0 = cy - 0.5f*h, bx1 = cx + 0.5f*w, by1 = cy + 0.5f*h;
        float area_b = (bx1 - bx0) * (by1 - by0);
        float bestov = -1.0f; int besti = 0;
        #pragma unroll
        for (int t = 0; t < T_; ++t) {
            float ax0 = st[t*5+0], ay0 = st[t*5+1], ax1 = st[t*5+2], ay1 = st[t*5+3];
            float ix0 = fmaxf(ax0, bx0), iy0 = fmaxf(ay0, by0);
            float ix1 = fminf(ax1, bx1), iy1 = fminf(ay1, by1);
            float iw = fmaxf(ix1 - ix0, 0.0f), ih = fmaxf(iy1 - iy0, 0.0f);
            float inter = iw * ih;
            float area_a = (ax1 - ax0) * (ay1 - ay0);
            float iou = inter / (area_a + area_b - inter);
            if (iou > bestov) { bestov = iou; besti = t; }   // first-max (strict >)
            // key: iou bits high (iou>=0, order-preserving) | ~p low -> ties pick smallest p
            unsigned long long key = ((unsigned long long)__float_as_uint(iou) << 32) |
                                     (unsigned long long)(0xFFFFFFFFu - (unsigned)p);
            atomicMax(&sbp[t], key);
        }
        bt_ov [(size_t)b * P_ + p] = bestov;
        bt_idx[(size_t)b * P_ + p] = besti;
    }
    __syncthreads();
    if (threadIdx.x < T_) atomicMax(&bp[b * T_ + threadIdx.x], sbp[threadIdx.x]);
}

// ---------------- K2: force each gt onto its best prior (sequential, last-wins) ----------------
__global__ void k2_force(const unsigned long long* __restrict__ bp,
                         float* __restrict__ bt_ov, int* __restrict__ bt_idx) {
    int b = threadIdx.x;
    if (b < B_) {
        for (int j = 0; j < T_; ++j) {
            unsigned pidx = 0xFFFFFFFFu - (unsigned)(bp[b * T_ + j] & 0xFFFFFFFFull);
            bt_ov [(size_t)b * P_ + pidx] = 2.0f;
            bt_idx[(size_t)b * P_ + pidx] = j;
        }
    }
}

// ---------------- K3: conf_t + SSD encode + smooth-L1 over positives ----------------
__global__ void __launch_bounds__(256)
k3_encode(const float* __restrict__ loc, const float* __restrict__ priors,
          const float* __restrict__ targets, const float* __restrict__ bt_ov,
          const int* __restrict__ bt_idx, int* __restrict__ conf_t,
          int* __restrict__ num_pos, float* __restrict__ loss_l_sum) {
    const int b = blockIdx.y;
    const int p = blockIdx.x * blockDim.x + threadIdx.x;
    float ll = 0.0f; int cnt = 0;
    if (p < P_) {
        float ov = bt_ov[(size_t)b * P_ + p];
        int   ti = bt_idx[(size_t)b * P_ + p];
        float lab = targets[(b * T_ + ti) * 5 + 4];
        int ct = (ov < 0.5f) ? 0 : ((int)lab + 1);
        conf_t[(size_t)b * P_ + p] = ct;
        if (ct > 0) {
            cnt = 1;
            float mx0 = targets[(b*T_+ti)*5+0], my0 = targets[(b*T_+ti)*5+1];
            float mx1 = targets[(b*T_+ti)*5+2], my1 = targets[(b*T_+ti)*5+3];
            float cx = priors[p*4+0], cy = priors[p*4+1], w = priors[p*4+2], h = priors[p*4+3];
            float g0 = ((mx0 + mx1) * 0.5f - cx) / (0.1f * w);
            float g1 = ((my0 + my1) * 0.5f - cy) / (0.1f * h);
            float g2 = logf((mx1 - mx0) / w) / 0.2f;
            float g3 = logf((my1 - my0) / h) / 0.2f;
            const float* lp = loc + ((size_t)b * P_ + p) * 4;
            ll = sl1(lp[0]-g0) + sl1(lp[1]-g1) + sl1(lp[2]-g2) + sl1(lp[3]-g3);
        }
    }
    #pragma unroll
    for (int off = 16; off; off >>= 1) { ll += __shfl_xor(ll, off); cnt += __shfl_xor(cnt, off); }
    __shared__ float sll; __shared__ int scnt;
    if (threadIdx.x == 0) { sll = 0.0f; scnt = 0; }
    __syncthreads();
    if ((threadIdx.x & 31) == 0) { atomicAdd(&sll, ll); atomicAdd(&scnt, cnt); }
    __syncthreads();
    if (threadIdx.x == 0) { atomicAdd(loss_l_sum, sll); atomicAdd(&num_pos[b], scnt); }
}

// ---------------- K4: streaming logsumexp via WMMA (the 518MB pass) ----------------
// One wave32 handles a 16-prior tile. Lane m (m<16) loads classes [g*32 .. g*32+15],
// lane m+16 loads [g*32+16 .. g*32+31] of row m -- contiguous k order (k = 16h+s); fine since B is all ones.
// D = A x ones accumulates 3 chained v_wmma_f32_16x16x32_bf16 -> per-row exp-sums.
// Bulk stream is non-temporal (518MB > 192MB L2) to protect the hot workspace.
__global__ void __launch_bounds__(256)
k4_conf(const float* __restrict__ conf, const int* __restrict__ conf_t,
        float* __restrict__ loss_c, float* __restrict__ cepart) {
    const int b    = blockIdx.y;
    const int wave = threadIdx.x >> 5;
    const int lane = threadIdx.x & 31;
    const int tile = blockIdx.x * 8 + wave;
    if (tile >= NTILES) return;
    const int base = tile * 16;
    const int half = lane >> 4;     // 0: K low half, 1: K high half
    const int mrow = lane & 15;     // matrix row within tile
    int row  = base + mrow;
    int rowc = row < P_ ? row : (P_ - 1);          // clamp; keeps EXEC full for WMMA
    const float* rp = conf + ((size_t)b * P_ + rowc) * NCLS;
    __builtin_prefetch(rp + 16 * NCLS, 0, 1);      // global_prefetch next tile's rows

    float x0[16], x1[16], x2[16];
    const int c0 = half * 16, c1 = 32 + half * 16;
    #pragma unroll
    for (int i = 0; i < 16; ++i) {
        x0[i] = __builtin_nontemporal_load(rp + c0 + i);
        x1[i] = __builtin_nontemporal_load(rp + c1 + i);
    }
    if (half == 0) {
        #pragma unroll
        for (int i = 0; i < 16; ++i)
            x2[i] = __builtin_nontemporal_load(rp + 64 + i);      // classes 64..79
    } else {
        x2[0] = __builtin_nontemporal_load(rp + 80);              // class 80 only
        #pragma unroll
        for (int i = 1; i < 16; ++i) x2[i] = -__builtin_inff();   // pad -> exp = 0
    }
    float pmax = -__builtin_inff();
    #pragma unroll
    for (int i = 0; i < 16; ++i) {
        pmax = fmaxf(pmax, x0[i]); pmax = fmaxf(pmax, x1[i]); pmax = fmaxf(pmax, x2[i]);
    }
    float rmax = fmaxf(pmax, __shfl_xor(pmax, 16));  // full row max (81 classes)

    v16us u0, u1, u2, uo;
    #pragma unroll
    for (int i = 0; i < 16; ++i) {
        u0[i] = f2bf_rne(__expf(x0[i] - rmax));
        u1[i] = f2bf_rne(__expf(x1[i] - rmax));
        u2[i] = f2bf_rne(__expf(x2[i] - rmax));
        uo[i] = 0x3F80u;                              // bf16 1.0 (ones B-matrix)
    }
    v16bf a0 = __builtin_bit_cast(v16bf, u0);
    v16bf a1 = __builtin_bit_cast(v16bf, u1);
    v16bf a2 = __builtin_bit_cast(v16bf, u2);
    v16bf on = __builtin_bit_cast(v16bf, uo);

    v8f c = {};
    c = __builtin_amdgcn_wmma_f32_16x16x32_bf16(false, a0, false, on, (short)0, c, false, false);
    c = __builtin_amdgcn_wmma_f32_16x16x32_bf16(false, a1, false, on, (short)0, c, false, false);
    c = __builtin_amdgcn_wmma_f32_16x16x32_bf16(false, a2, false, on, (short)0, c, false, false);
    // C/D layout: VGPR r of lanes 0-15 = row r; of lanes 16-31 = row r+8 (all columns equal).

    float cepos = 0.0f;
    #pragma unroll
    for (int r = 0; r < 8; ++r) {
        float shi = __shfl_xor(c[r], 16);            // lane r+8 <- lane r+24: row r+8 sum
        if (lane < 16) {
            int myrow = -1; float S = 0.0f;
            if      (mrow == r)     { myrow = base + r;     S = c[r]; }
            else if (mrow == r + 8) { myrow = base + r + 8; S = shi;  }
            if (myrow >= 0 && myrow < P_) {          // owner lane also holds rmax of its row
                float lse = __logf(S) + rmax;
                int   ct  = conf_t[(size_t)b * P_ + myrow];
                float cg  = conf[((size_t)b * P_ + myrow) * NCLS + ct];
                float ce  = lse - cg;
                loss_c[(size_t)b * P_ + myrow] = (ct > 0) ? 0.0f : ce;
                if (ct > 0) cepos += ce;
            }
        }
    }
    #pragma unroll
    for (int off = 16; off; off >>= 1) cepos += __shfl_xor(cepos, off);
    if (lane == 0) cepart[(size_t)b * NTILES + tile] = cepos;   // deterministic partials
}

// ---------------- K5: top-k sum; TDM stages loss_c[b] (100KB) into LDS once ----------------
// 31-step binary search on float bits finds the k-th largest exactly; the selected-set
// SUM is tie-invariant, so this reproduces the double-argsort mining result exactly.
// All 32 scan passes then run out of LDS instead of L2.
__global__ void __launch_bounds__(256)
k5_mine(const float* __restrict__ loss_c, const int* __restrict__ num_pos,
        float* __restrict__ negsum) {
    const int b = blockIdx.x;
    const float* lc = loss_c + (size_t)b * P_;
    __shared__ float sl[P_];                         // 100 KB of the 320 KB WGP LDS
    __shared__ int scnt; __shared__ float ssum;
    int k = 3 * num_pos[b]; if (k > P_ - 1) k = P_ - 1;
    if (k <= 0) { if (threadIdx.x == 0) negsum[b] = 0.0f; return; }

    if (threadIdx.x == 0) {
        // Tensor DMA descriptor (ISA 08 §8): 1-row 2D tile, 25000 x 4B elements.
        unsigned long long ga = (unsigned long long)(const void*)lc;
        unsigned ldsa = (unsigned)(unsigned long long)(void*)sl;   // low 32b = LDS offset
        u32x4 g0;
        g0[0] = 1u;                                              // count=1 (valid user D#)
        g0[1] = ldsa;                                            // lds_addr
        g0[2] = (unsigned)(ga & 0xFFFFFFFFull);                  // global_addr[31:0]
        g0[3] = (unsigned)((ga >> 32) & 0x01FFFFFFull) | (2u << 30); // addr[56:32] | type=2
        i32x8 g1;
        g1[0] = 0x00020000;                                      // data_size=2 -> 4 bytes
        g1[1] = (int)(((unsigned)P_ & 0xFFFFu) << 16);           // tensor_dim0[15:0]
        g1[2] = (int)((((unsigned)P_ >> 16) & 0xFFFFu) | (1u << 16)); // dim0 hi | tensor_dim1=1
        g1[3] = (int)(((unsigned)P_ & 0xFFFFu) << 16);           // tile_dim0 = 25000
        g1[4] = 1;                                               // tile_dim1 = 1
        g1[5] = P_;                                              // tensor_dim0_stride lo
        g1[6] = 0; g1[7] = 0;
        i32x4 gz4 = {0, 0, 0, 0};                                // groups 2/3 unused (<=2D)
        i32x8 gz8 = {0, 0, 0, 0, 0, 0, 0, 0};                    // 6-arg form (clang-23)
        __builtin_amdgcn_tensor_load_to_lds(g0, g1, gz4, gz4, gz8, 0);
        __builtin_amdgcn_s_wait_tensorcnt(0);
    }
    __syncthreads();                                             // publish LDS tile

    unsigned lo = 0u, hi = 0x7F800000u;              // loss_c >= 0, finite
    for (int it = 0; it < 31; ++it) {
        unsigned mid = lo + ((hi - lo) >> 1);
        if (threadIdx.x == 0) scnt = 0;
        __syncthreads();
        int c = 0;
        for (int p = threadIdx.x; p < P_; p += blockDim.x)
            if (__float_as_uint(sl[p]) >= mid) c++;
        atomicAdd(&scnt, c);
        __syncthreads();
        if (scnt >= k) lo = mid; else hi = mid;
        __syncthreads();
    }
    if (threadIdx.x == 0) { scnt = 0; ssum = 0.0f; }
    __syncthreads();
    int cgt = 0; float sgt = 0.0f;
    for (int p = threadIdx.x; p < P_; p += blockDim.x) {
        float v = sl[p];
        if (__float_as_uint(v) > lo) { cgt++; sgt += v; }
    }
    atomicAdd(&scnt, cgt); atomicAdd(&ssum, sgt);
    __syncthreads();
    if (threadIdx.x == 0)
        negsum[b] = ssum + (float)(k - scnt) * __uint_as_float(lo);  // tie-invariant top-k sum
}

// ---------------- K6: final combine ----------------
__global__ void __launch_bounds__(256)
k6_final(const float* __restrict__ cepart, const int* __restrict__ num_pos,
         const float* __restrict__ negsum, const float* __restrict__ scal,
         float* __restrict__ out) {
    __shared__ float red[256];
    float s = 0.0f;
    for (int i = threadIdx.x; i < B_ * NTILES; i += blockDim.x) s += cepart[i];
    red[threadIdx.x] = s;
    __syncthreads();
    for (int off = 128; off; off >>= 1) {
        if (threadIdx.x < off) red[threadIdx.x] += red[threadIdx.x + off];
        __syncthreads();
    }
    if (threadIdx.x == 0) {
        float cetot = red[0];
        int np = 0; float ng = 0.0f;
        for (int b = 0; b < B_; ++b) { np += num_pos[b]; ng += negsum[b]; }
        float N = (float)np; if (N < 1.0f) N = 1.0f;
        out[0] = scal[0] / N;            // loss_l / N
        out[1] = (cetot + ng) / N;       // loss_conf / N
    }
}

extern "C" void kernel_launch(void* const* d_in, const int* in_sizes, int n_in,
                              void* d_out, int out_size, void* d_ws, size_t ws_size,
                              hipStream_t stream) {
    const float* loc     = (const float*)d_in[0];   // [64,25000,4]
    const float* conf    = (const float*)d_in[1];   // [64,25000,81]
    const float* priors  = (const float*)d_in[2];   // [25000,4]
    const float* targets = (const float*)d_in[3];   // [64,16,5]
    float* out = (float*)d_out;

    char* w = (char*)d_ws;
    const size_t BP4 = (size_t)B_ * P_ * 4;         // 6.4 MB, 256B-aligned
    float* bt_ov  = (float*)(w + 0 * BP4);
    int*   bt_idx = (int*)  (w + 1 * BP4);
    int*   conf_t = (int*)  (w + 2 * BP4);
    float* loss_c = (float*)(w + 3 * BP4);
    unsigned long long* bp = (unsigned long long*)(w + 4 * BP4);           // 8 KB
    int*   num_pos = (int*)  (w + 4 * BP4 + 8192);
    float* negsum  = (float*)(w + 4 * BP4 + 8192 + 256);
    float* scal    = (float*)(w + 4 * BP4 + 8192 + 512);                   // [loss_l, spare]
    float* cepart  = (float*)(w + 4 * BP4 + 8192 + 768);                   // B*NTILES ~ 400 KB

    k0_init<<<4, 256, 0, stream>>>(bp, num_pos, negsum, scal);
    dim3 g1((P_ + 255) / 256, B_);
    k1_match<<<g1, 256, 0, stream>>>(priors, targets, bt_ov, bt_idx, bp);
    k2_force<<<1, 64, 0, stream>>>(bp, bt_ov, bt_idx);
    k3_encode<<<g1, 256, 0, stream>>>(loc, priors, targets, bt_ov, bt_idx,
                                      conf_t, num_pos, scal);
    dim3 g4((NTILES + 7) / 8, B_);
    k4_conf<<<g4, 256, 0, stream>>>(conf, conf_t, loss_c, cepart);
    k5_mine<<<B_, 256, 0, stream>>>(loss_c, num_pos, negsum);
    k6_final<<<1, 256, 0, stream>>>(cepart, num_pos, negsum, scal, out);
}
